// DiagSSMBlock_82523501625910
// MI455X (gfx1250) — matrix operand order repaired and checked by reading, not compiled, not run on hardware
//
#include <hip/hip_runtime.h>

// ---------------------------------------------------------------------------
// DiagSSM: out[t,h] = sum_{k<=t} a[h]^(t-k) * (X @ B)[k,h]
//   Stage 1: S = X @ B   (WMMA bf16x3 split fp32 GEMM, fp32 accumulate,
//                         double-buffered LDS, fragment-major tile layouts)
//   Stage 2: chunked scan y[t] = a*y[t-1] + S[t]  (3 bandwidth-bound passes)
// ---------------------------------------------------------------------------

#define T_DIM 4096
#define H_DIM 2048

#define BM 128
#define BN 128
#define BK 32
#define NKB (H_DIM / BK)       // 64 K-blocks
#define LDA (BK + 8)           // A row stride (elems): 80 B, 16B-aligned
#define LDB (BK + 8)           // B stored fragment-major [col][k]

#define NCHUNK 64
#define CLEN   (T_DIM / NCHUNK)   // 64

typedef __attribute__((ext_vector_type(16))) __bf16 v16bf;
typedef __attribute__((ext_vector_type(8)))  float  v8f;

__device__ __forceinline__ void split_bf16(float x, __bf16& hi, __bf16& lo) {
    hi = (__bf16)x;
    lo = (__bf16)(x - (float)hi);
}

// ---------------------------------------------------------------------------
// S = X (T,H) @ B (H,H) -> S (T,H).
// 256 threads = 8 waves; WG tile 128x128; wave tile 32x64 = 2x4 WMMA tiles.
// ---------------------------------------------------------------------------
__global__ __launch_bounds__(256)
void gemm_xb_bf16x3(const float* __restrict__ X,
                    const float* __restrict__ B,
                    float* __restrict__ S)
{
    // double-buffered tiles, fragment-friendly layouts
    __shared__ __align__(16) __bf16 sAhi[2][BM][LDA];   // [row][k]
    __shared__ __align__(16) __bf16 sAlo[2][BM][LDA];
    __shared__ __align__(16) __bf16 sBhi[2][BN][LDB];   // [col][k]  (transposed)
    __shared__ __align__(16) __bf16 sBlo[2][BN][LDB];

    const int tid     = threadIdx.x;
    const int rowBase = blockIdx.y * BM;
    const int colBase = blockIdx.x * BN;

    const int wid   = tid >> 5;
    const int lane  = tid & 31;
    const int wm    = wid >> 1;      // 0..3 : 32-row slab
    const int wn    = wid & 1;       // 0..1 : 64-col slab
    const int mlane = lane & 15;
    const int half  = lane >> 4;

    // cooperative-fill coordinates
    const int ar   = tid >> 1;          // 0..127  (A row)
    const int ac   = (tid & 1) * 16;    // 0 or 16 (A k-segment)
    const int bcol = tid & 127;         // 0..127  (B column)
    const int bks  = (tid >> 7) * 16;   // 0 or 16 (B k-segment)

    const float* gA = X + (size_t)(rowBase + ar) * H_DIM + ac;          // + k0
    const float* gB = B + (size_t)bks * H_DIM + colBase + bcol;         // + k0*H

    v8f acc[2][4];
    {
        v8f z = {};
        for (int i = 0; i < 2; ++i)
            for (int j = 0; j < 4; ++j) acc[i][j] = z;
    }

    float pa[16], pb[16];

    // ---- prologue: load + store K-block 0 into buffer 0 ----
    #pragma unroll
    for (int i = 0; i < 16; i += 4) {
        float4 v = *(const float4*)(gA + i);
        pa[i] = v.x; pa[i+1] = v.y; pa[i+2] = v.z; pa[i+3] = v.w;
    }
    #pragma unroll
    for (int j = 0; j < 16; ++j) pb[j] = gB[(size_t)j * H_DIM];
    #pragma unroll
    for (int i = 0; i < 16; ++i) {
        split_bf16(pa[i], sAhi[0][ar][ac + i],    sAlo[0][ar][ac + i]);
        split_bf16(pb[i], sBhi[0][bcol][bks + i], sBlo[0][bcol][bks + i]);
    }
    __syncthreads();

    for (int kb = 0; kb < NKB; ++kb) {
        const int cur = kb & 1;
        const bool more = (kb + 1) < NKB;

        // ---- issue global loads for next K-block (latency hidden by WMMAs) --
        if (more) {
            const int k0n = (kb + 1) * BK;
            #pragma unroll
            for (int i = 0; i < 16; i += 4) {
                float4 v = *(const float4*)(gA + k0n + i);
                pa[i] = v.x; pa[i+1] = v.y; pa[i+2] = v.z; pa[i+3] = v.w;
            }
            #pragma unroll
            for (int j = 0; j < 16; ++j)
                pb[j] = gB[(size_t)(k0n) * H_DIM + (size_t)j * H_DIM];
        }
        if (kb + 2 < NKB) {   // warm L2 for the K+2 tile (global_prefetch_b8)
            const int k0p = (kb + 2) * BK;
            __builtin_prefetch(gA + k0p, 0, 3);
            __builtin_prefetch(gB + (size_t)k0p * H_DIM, 0, 3);
        }

        // ---- A fragments: 16x32 bf16 layout ----
        v16bf aHi[2], aLo[2];
        #pragma unroll
        for (int mt = 0; mt < 2; ++mt) {
            const int row = wm * 32 + mt * 16 + mlane;
            #pragma unroll
            for (int i = 0; i < 8; ++i) {
                const int kk = half * 16 + i;
                aHi[mt][i]     = sAhi[cur][row][kk];
                aLo[mt][i]     = sAlo[cur][row][kk];
                aHi[mt][8 + i] = sAhi[cur][row][8 + kk];
                aLo[mt][8 + i] = sAlo[cur][row][8 + kk];
            }
        }

        // ---- B fragments (contiguous 32 B per lane) + 3 WMMAs per tile ----
        #pragma unroll
        for (int nt = 0; nt < 4; ++nt) {
            const int col = wn * 64 + nt * 16 + mlane;
            v16bf bHi, bLo;
            #pragma unroll
            for (int i = 0; i < 16; ++i) {
                const int kk = half * 16 + i;
                bHi[i] = sBhi[cur][col][kk];
                bLo[i] = sBlo[cur][col][kk];
            }
            #pragma unroll
            for (int mt = 0; mt < 2; ++mt) {
                acc[mt][nt] = __builtin_amdgcn_wmma_f32_16x16x32_bf16(
                    false, aHi[mt], false, bHi, (short)0, acc[mt][nt], false, false);
                acc[mt][nt] = __builtin_amdgcn_wmma_f32_16x16x32_bf16(
                    false, aHi[mt], false, bLo, (short)0, acc[mt][nt], false, false);
                acc[mt][nt] = __builtin_amdgcn_wmma_f32_16x16x32_bf16(
                    false, aLo[mt], false, bHi, (short)0, acc[mt][nt], false, false);
            }
        }

        // ---- convert + store prefetched data into the other buffer ----
        if (more) {
            const int nxt = cur ^ 1;
            #pragma unroll
            for (int i = 0; i < 16; ++i) {
                split_bf16(pa[i], sAhi[nxt][ar][ac + i],    sAlo[nxt][ar][ac + i]);
                split_bf16(pb[i], sBhi[nxt][bcol][bks + i], sBlo[nxt][bcol][bks + i]);
            }
        }
        __syncthreads();
    }

    // ---- store C/D tiles: VGPR v -> M = half*8 + v, N = lane&15 ----
    #pragma unroll
    for (int mt = 0; mt < 2; ++mt) {
        #pragma unroll
        for (int nt = 0; nt < 4; ++nt) {
            const int row0 = rowBase + wm * 32 + mt * 16 + half * 8;
            const int col  = colBase + wn * 64 + nt * 16 + mlane;
            #pragma unroll
            for (int v = 0; v < 8; ++v)
                S[(size_t)(row0 + v) * H_DIM + col] = acc[mt][nt][v];
        }
    }
}

// ---------------------------------------------------------------------------
// Chunked diagonal scan: y[t] = a*y[t-1] + S[t], NCHUNK chunks of CLEN.
// ---------------------------------------------------------------------------
__global__ __launch_bounds__(256)
void scan_local(const float* __restrict__ S, const float* __restrict__ a_diag,
                float* __restrict__ carry)
{
    const int h = blockIdx.x * blockDim.x + threadIdx.x;
    const int c = blockIdx.y;
    const float a = a_diag[h];
    float y = 0.0f;
    const float* p = S + (size_t)c * CLEN * H_DIM + h;
    #pragma unroll 4
    for (int k = 0; k < CLEN; ++k)
        y = fmaf(a, y, p[(size_t)k * H_DIM]);
    carry[c * H_DIM + h] = y;
}

__global__ __launch_bounds__(256)
void scan_prefix(const float* __restrict__ carry, const float* __restrict__ a_diag,
                 float* __restrict__ prefix)
{
    const int h = blockIdx.x * blockDim.x + threadIdx.x;
    const float a = a_diag[h];
    float aL = a;
    #pragma unroll
    for (int i = 0; i < 6; ++i) aL *= aL;   // a^64 == a^CLEN
    float p = 0.0f;
    for (int c = 0; c < NCHUNK; ++c) {
        prefix[c * H_DIM + h] = p;
        p = fmaf(aL, p, carry[c * H_DIM + h]);
    }
}

__global__ __launch_bounds__(256)
void scan_final(const float* __restrict__ S, const float* __restrict__ a_diag,
                const float* __restrict__ prefix, float* __restrict__ out)
{
    const int h = blockIdx.x * blockDim.x + threadIdx.x;
    const int c = blockIdx.y;
    const float a = a_diag[h];
    float y = prefix[c * H_DIM + h];
    const float* p = S   + (size_t)c * CLEN * H_DIM + h;
    float*       q = out + (size_t)c * CLEN * H_DIM + h;
    #pragma unroll 4
    for (int k = 0; k < CLEN; ++k) {
        y = fmaf(a, y, p[(size_t)k * H_DIM]);
        q[(size_t)k * H_DIM] = y;
    }
}

// ---------------------------------------------------------------------------
extern "C" void kernel_launch(void* const* d_in, const int* in_sizes, int n_in,
                              void* d_out, int out_size, void* d_ws, size_t ws_size,
                              hipStream_t stream)
{
    const float* x_seq  = (const float*)d_in[0];  // (T,H)
    const float* a_diag = (const float*)d_in[1];  // (H,)
    const float* b_mat  = (const float*)d_in[2];  // (H,H)
    float* out = (float*)d_out;                   // (T,H)

    float* S      = (float*)d_ws;                          // T*H floats (32 MB)
    float* carry  = S + (size_t)T_DIM * H_DIM;             // NCHUNK*H floats
    float* prefix = carry + (size_t)NCHUNK * H_DIM;        // NCHUNK*H floats

    dim3 gGemm(H_DIM / BN, T_DIM / BM);   // 16 x 32 workgroups
    gemm_xb_bf16x3<<<gGemm, 256, 0, stream>>>(x_seq, b_mat, S);

    dim3 gScan(H_DIM / 256, NCHUNK);      // 8 x 64
    scan_local<<<gScan, 256, 0, stream>>>(S, a_diag, carry);
    scan_prefix<<<dim3(H_DIM / 256), 256, 0, stream>>>(carry, a_diag, prefix);
    scan_final<<<gScan, 256, 0, stream>>>(S, a_diag, prefix, out);
}
